// My_model_drug_4277787427386
// MI455X (gfx1250) — matrix operand order repaired and checked by reading, not compiled, not run on hardware
//
#include <hip/hip_runtime.h>
#include <hip/hip_bf16.h>
#include <math.h>

// ---------------- types for WMMA ----------------
typedef __attribute__((ext_vector_type(16))) __bf16 v16bf;
typedef __attribute__((ext_vector_type(8)))  float  v8f;

__device__ __forceinline__ unsigned short f2bf(float x) {
    unsigned u = __float_as_uint(x);
    u += 0x7FFFu + ((u >> 16) & 1u);          // round-to-nearest-even
    return (unsigned short)(u >> 16);
}
__device__ __forceinline__ unsigned pk2(float a, float b) {
    return (unsigned)f2bf(a) | ((unsigned)f2bf(b) << 16);
}

// order-preserving float<->uint encoding (for atomic max over signed floats)
__device__ __forceinline__ unsigned fenc(float f) {
    unsigned u = __float_as_uint(f);
    return (u & 0x80000000u) ? ~u : (u | 0x80000000u);
}
__device__ __forceinline__ float fdec(unsigned e) {
    return __uint_as_float((e & 0x80000000u) ? (e ^ 0x80000000u) : ~e);
}

// ---------------- problem constants ----------------
#define BSZ   512
#define NDpg  40
#define NDRUG (BSZ*NDpg)      // 20480
#define N0c   706
#define NCELL (BSZ*N0c)       // 361472
#define NC1   (BSZ*512)       // 262144
#define NC2   (BSZ*400)       // 204800
#define EDRUG (BSZ*160)       // 81920
#define EC0   (BSZ*8000)      // 4096000
#define BN_EPS 1e-5f

// ======================================================================
// bf16-WMMA GEMM, LDS-staged: C = act(A(MxK) @ B(KxN) + bias)
// 128 threads = 4 waves (2x2 wave tiles) -> 32x32 block tile.
// REQUIRES (enforced by the launcher): M%32==0, N%32==0, K%32==0,
// lda%4==0, ldb%4==0. Zero masking anywhere -> no divergence; EXEC is
// all-ones through v_wmma. ACT: 0=none 1=relu 2=elu
// ======================================================================
template<int ACT>
__global__ __launch_bounds__(128)
void wmma_gemm(const float* __restrict__ A, int lda,
               const float* __restrict__ Bw, int ldb,
               const float* __restrict__ bias,
               float* __restrict__ C, int ldc,
               int M, int N, int K)
{
    __shared__ __align__(16) unsigned short As[32 * 32];   // [row][k]
    __shared__ __align__(16) unsigned short Bs[32 * 32];   // [n][k] (transposed)

    const int tid   = threadIdx.x;
    const int tileM = blockIdx.x * 32;
    const int tileN = blockIdx.y * 32;
    const int wave  = tid >> 5;
    const int lane  = tid & 31;
    const int hi    = lane >> 4;
    const int l16   = lane & 15;
    const int wm    = wave & 1;
    const int wn    = wave >> 1;
    const int arow  = tid >> 2;               // 0..31
    const int akb   = (tid & 3) * 8;          // 0,8,16,24

    v8f acc = {};
    const float* aprow = A + (size_t)(tileM + arow) * lda + akb;
    for (int k0 = 0; k0 < K; k0 += 32) {
        // ---- A: 8 contiguous floats/thread via 2x b128, pack, 1x ds_store_b128
        const float4* ap = (const float4*)(aprow + k0);
        float4 u0 = ap[0], u1 = ap[1];
        uint4 pk;
        pk.x = pk2(u0.x, u0.y); pk.y = pk2(u0.z, u0.w);
        pk.z = pk2(u1.x, u1.y); pk.w = pk2(u1.z, u1.w);
        *(uint4*)&As[arow * 32 + akb] = pk;
        // ---- B: float4 over n (coalesced), transposed b16 stores
        #pragma unroll
        for (int j = 0; j < 2; ++j) {
            int idx = tid + j * 128;              // 0..255
            int kk = idx >> 3, nb = (idx & 7) * 4;
            float4 w = *(const float4*)(Bw + (size_t)(k0 + kk) * ldb + tileN + nb);
            Bs[(nb + 0) * 32 + kk] = f2bf(w.x);
            Bs[(nb + 1) * 32 + kk] = f2bf(w.y);
            Bs[(nb + 2) * 32 + kk] = f2bf(w.z);
            Bs[(nb + 3) * 32 + kk] = f2bf(w.w);
        }
        __syncthreads();
        // ---- fragments (ISA 7.12.2): A k-runs [hi*8,+8) & [16+hi*8,+8);
        //      B k-runs [hi*8,+8) & [16+hi*8,+8) from transposed tile
        union { v16bf v; uint4 q[2]; } af, bfr;
        const unsigned short* ar = &As[(wm * 16 + l16) * 32];
        const unsigned short* bc = &Bs[(wn * 16 + l16) * 32];
        af.q[0]  = *(const uint4*)&ar[hi * 8];
        af.q[1]  = *(const uint4*)&ar[16 + hi * 8];
        bfr.q[0] = *(const uint4*)&bc[hi * 8];
        bfr.q[1] = *(const uint4*)&bc[16 + hi * 8];
        acc = __builtin_amdgcn_wmma_f32_16x16x32_bf16(
                  false, af.v, false, bfr.v, (short)0, acc, false, false);
        __syncthreads();
    }
    // ---- store: VGPR r -> row = r + 8*hi within wave tile (always in-bounds)
    #pragma unroll
    for (int r = 0; r < 8; ++r) {
        int row = tileM + wm * 16 + r + 8 * hi;
        int col = tileN + wn * 16 + l16;
        float v = acc[r];
        v += bias[col];
        if (ACT == 1) v = fmaxf(v, 0.0f);
        else if (ACT == 2) v = (v > 0.0f) ? v : (expf(v) - 1.0f);
        C[(size_t)row * ldc + col] = v;
    }
}

// regression head: out[row] = dot(x[row,0:384], w) + b  (matvec, 1 wave/row)
__global__ __launch_bounds__(32)
void head_dot_k(const float* __restrict__ x, const float* __restrict__ w,
                const float* __restrict__ b, float* __restrict__ out)
{
    int row = blockIdx.x;
    int lane = threadIdx.x;
    const float* xr = x + (size_t)row * 384;
    float s = 0.f;
    #pragma unroll
    for (int c = 0; c < 384; c += 32) s += xr[c + lane] * w[c + lane];
    #pragma unroll
    for (int off = 16; off > 0; off >>= 1) s += __shfl_xor(s, off, 32);
    if (lane == 0) out[row] = s + b[0];
}

// ======================================================================
// elementwise / graph kernels
// ======================================================================
__global__ void scatter_add_k(const float* __restrict__ x, int ldx,
                              const int* __restrict__ src, const int* __restrict__ dst,
                              float* __restrict__ agg, int ldagg, int F, int E)
{
    long i = (long)blockIdx.x * blockDim.x + threadIdx.x;
    if (i >= (long)E * F) return;
    int e = (int)(i / F), f = (int)(i % F);
    atomicAdd(&agg[(size_t)dst[e] * ldagg + f], x[(size_t)src[e] * ldx + f]);
}

__global__ void add2_k(float* __restrict__ y, int ldy,
                       const float* __restrict__ x, int ldx, int rows, int cols)
{
    long i = (long)blockIdx.x * blockDim.x + threadIdx.x;
    if (i >= (long)rows * cols) return;
    int r = (int)(i / cols), c = (int)(i % cols);
    y[(size_t)r * ldy + c] += x[(size_t)r * ldx + c];
}

__global__ void bn_stats_k(const float* __restrict__ x, int ldx, int rows,
                           float* __restrict__ mu, float* __restrict__ rs)
{
    int c = blockIdx.x;
    __shared__ float ssum[256], ssq[256];
    float s = 0.f, q = 0.f;
    for (int r = threadIdx.x; r < rows; r += blockDim.x) {
        float v = x[(size_t)r * ldx + c];
        s += v; q += v * v;
    }
    ssum[threadIdx.x] = s; ssq[threadIdx.x] = q;
    __syncthreads();
    for (int off = blockDim.x >> 1; off > 0; off >>= 1) {
        if (threadIdx.x < (unsigned)off) {
            ssum[threadIdx.x] += ssum[threadIdx.x + off];
            ssq[threadIdx.x]  += ssq[threadIdx.x + off];
        }
        __syncthreads();
    }
    if (threadIdx.x == 0) {
        float m = ssum[0] / rows;
        float var = ssq[0] / rows - m * m;
        mu[c] = m;
        rs[c] = rsqrtf(var + BN_EPS);
    }
}

__global__ void bn_apply_k(const float* __restrict__ x, int ldx, int rows, int cols,
                           const float* __restrict__ mu, const float* __restrict__ rs,
                           const float* __restrict__ gamma, const float* __restrict__ beta,
                           float* __restrict__ out, int ldo)
{
    long i = (long)blockIdx.x * blockDim.x + threadIdx.x;
    if (i >= (long)rows * cols) return;
    int r = (int)(i / cols), c = (int)(i % cols);
    float v = (x[(size_t)r * ldx + c] - mu[c]) * rs[c];
    if (gamma) v = v * gamma[c] + beta[c];
    out[(size_t)r * ldo + c] = v;
}

__global__ void segmax_drug_k(const float* __restrict__ rep, float* __restrict__ xd)
{
    int i = blockIdx.x * blockDim.x + threadIdx.x;
    if (i >= BSZ * 384) return;
    int g = i / 384, c = i % 384;
    float m = -INFINITY;
    for (int r = 0; r < NDpg; ++r)
        m = fmaxf(m, rep[(size_t)(g * NDpg + r) * 384 + c]);
    xd[i] = m;
}

// per-node h = x @ W (Kx16), plus attention scores hs,hd
__global__ void gat_h_k(const float* __restrict__ x, int K,
                        const float* __restrict__ W,
                        const float* __restrict__ as_, const float* __restrict__ ad_,
                        float* __restrict__ h, float* __restrict__ hs,
                        float* __restrict__ hd, int n)
{
    int i = blockIdx.x * blockDim.x + threadIdx.x;
    if (i >= n) return;
    float acc[16];
    #pragma unroll
    for (int c = 0; c < 16; ++c) acc[c] = 0.f;
    for (int k = 0; k < K; ++k) {
        float xv = x[(size_t)i * K + k];
        #pragma unroll
        for (int c = 0; c < 16; ++c) acc[c] += xv * W[k * 16 + c];
    }
    float s = 0.f, d = 0.f;
    #pragma unroll
    for (int c = 0; c < 16; ++c) {
        h[(size_t)i * 16 + c] = acc[c];
        s += acc[c] * as_[c];
        d += acc[c] * ad_[c];
    }
    hs[i] = s; hd[i] = d;
}

// pass 1: leaky-relu logits + encoded segment max over dst (self-loops appended)
__global__ void gat_logit_max_k(const float* __restrict__ hs, const float* __restrict__ hd,
                                const int* __restrict__ src, const int* __restrict__ dst,
                                int E, int n, float* __restrict__ ev, unsigned* __restrict__ m)
{
    int i = blockIdx.x * blockDim.x + threadIdx.x;
    if (i >= E + n) return;
    int sN = (i < E) ? src[i] : (i - E);
    int dN = (i < E) ? dst[i] : (i - E);
    float l = hs[sN] + hd[dN];
    l = (l > 0.f) ? l : 0.2f * l;
    ev[i] = l;
    atomicMax(&m[dN], fenc(l));
}

// pass 2: e = exp(logit - max[dst]); segment sum
__global__ void gat_exp_sum_k(float* __restrict__ ev, const int* __restrict__ dst,
                              int E, int n, const unsigned* __restrict__ m,
                              float* __restrict__ s)
{
    int i = blockIdx.x * blockDim.x + threadIdx.x;
    if (i >= E + n) return;
    int dN = (i < E) ? dst[i] : (i - E);
    float e = expf(ev[i] - fdec(m[dN]));
    ev[i] = e;
    atomicAdd(&s[dN], e);
}

// pass 3: out[dst] += h[src] * (e / s[dst])
__global__ void gat_accum_k(const float* __restrict__ ev, const float* __restrict__ h,
                            const int* __restrict__ src, const int* __restrict__ dst,
                            const float* __restrict__ s, float* __restrict__ out,
                            int E, int n)
{
    long i = (long)blockIdx.x * blockDim.x + threadIdx.x;
    long tot = ((long)E + n) * 16;
    if (i >= tot) return;
    int idx = (int)(i >> 4), c = (int)(i & 15);
    int sN = (idx < E) ? src[idx] : (idx - E);
    int dN = (idx < E) ? dst[idx] : (idx - E);
    float alpha = ev[idx] / s[dN];
    atomicAdd(&out[(size_t)dN * 16 + c], h[(size_t)sN * 16 + c] * alpha);
}

// pass 4: v = relu(out + bias), cluster max-pool (non-negative -> uint-bit max)
__global__ void gat_finish_pool_k(const float* __restrict__ out, const float* __restrict__ bias,
                                  const int* __restrict__ cluster,
                                  unsigned* __restrict__ pool, int n)
{
    long i = (long)blockIdx.x * blockDim.x + threadIdx.x;
    if (i >= (long)n * 16) return;
    int node = (int)(i >> 4), c = (int)(i & 15);
    float v = fmaxf(out[i] + bias[c], 0.0f);
    atomicMax(&pool[(size_t)cluster[node] * 16 + c], __float_as_uint(v));
}

__global__ void concat_k(const float* __restrict__ a, const float* __restrict__ b,
                         float* __restrict__ out)
{
    int i = blockIdx.x * blockDim.x + threadIdx.x;
    if (i >= BSZ * 384) return;
    int r = i / 384, c = i % 384;
    out[i] = (c < 128) ? a[r * 128 + c] : b[r * 256 + (c - 128)];
}

// ======================================================================
static inline int nblk(long n, int t) { return (int)((n + t - 1) / t); }

extern "C" void kernel_launch(void* const* d_in, const int* in_sizes, int n_in,
                              void* d_out, int out_size, void* d_ws, size_t ws_size,
                              hipStream_t stream)
{
    auto F = [&](int i) { return (const float*)d_in[i]; };
    auto I = [&](int i) { return (const int*)d_in[i]; };

    const float* drug_x = F(0);
    const float* cell_x = F(1);
    const float* demb_W = F(20); const float* demb_b = F(21);
    const float* cemb_W1 = F(30); const float* cemb_b1 = F(31);
    const float* cemb_W2 = F(32); const float* cemb_b2 = F(33);
    const float* reg_W1 = F(34); const float* reg_b1 = F(35);
    const float* reg_W2 = F(36); const float* reg_b2 = F(37);
    const float* reg_W3 = F(38); const float* reg_b3 = F(39);
    const int* dei = I(40);                 // (2, EDRUG)
    const int* ei0 = I(42);                 // (2, EC0)
    const int* ei1 = I(43);                 // (2, E1)
    const int* cl0 = I(44);                 // NCELL -> NC1
    const int* cl1 = I(45);                 // NC1   -> NC2
    const int  E1  = in_sizes[43] / 2;

    float* ws = (float*)d_ws;
    size_t o = 0;
    auto alloc = [&](size_t nf) { size_t r = o; o += nf; return r; };
    float* xd2  = ws + alloc((size_t)BSZ * 128);
    float* xcf  = ws + alloc((size_t)NC2 * 16);     // final cell BN output (512 x 6400)
    float* mu   = ws + alloc(128);
    float* rs   = ws + alloc(128);
    float* w1p  = ws + alloc((size_t)96 * 128);     // zero-padded gin0_W1 (96x128)
    float* a1   = ws + alloc((size_t)BSZ * 1024);
    float* a2   = ws + alloc((size_t)BSZ * 256);
    float* hcat = ws + alloc((size_t)BSZ * 384);
    float* r1   = ws + alloc((size_t)BSZ * 384);
    float* r2   = ws + alloc((size_t)BSZ * 384);
    float* temp = ws + o;                           // reused arena

    dim3 blk(128);
    auto ggrid = [](int M, int N) { return dim3((M + 31) / 32, (N + 31) / 32); };

    // ---------------- drug branch: 3x GIN + pool + embed ----------------
    {
        float* t_y   = temp;
        float* t_h1  = t_y  + (size_t)NDRUG * 128;
        float* t_x   = t_h1 + (size_t)NDRUG * 128;
        float* t_rep = t_x  + (size_t)NDRUG * 128;
        float* t_xd  = t_rep + (size_t)NDRUG * 384;

        const float* xin = drug_x; int ldin = 77, Fin = 77;
        for (int l = 0; l < 3; ++l) {
            const float* W1 = F(2 + 6 * l), *b1 = F(3 + 6 * l);
            const float* W2 = F(4 + 6 * l), *b2 = F(5 + 6 * l);
            const float* gg = F(6 + 6 * l), *bb = F(7 + 6 * l);

            // pad layer-0 (K=77) up to K=96 so everything is interior
            int ldy = (l == 0) ? 96 : 128;
            int Kg  = (l == 0) ? 96 : 128;
            const float* Wuse = W1;
            if (l == 0) {
                hipMemsetAsync(w1p, 0, (size_t)96 * 128 * sizeof(float), stream);
                hipMemcpyAsync(w1p, W1, (size_t)77 * 128 * sizeof(float),
                               hipMemcpyDeviceToDevice, stream);
                Wuse = w1p;
            }

            hipMemsetAsync(t_y, 0, (size_t)NDRUG * ldy * sizeof(float), stream);
            scatter_add_k<<<nblk((long)EDRUG * Fin, 256), 256, 0, stream>>>(
                xin, ldin, dei, dei + EDRUG, t_y, ldy, Fin, EDRUG);
            add2_k<<<nblk((long)NDRUG * Fin, 256), 256, 0, stream>>>(
                t_y, ldy, xin, ldin, NDRUG, Fin);

            wmma_gemm<1><<<ggrid(NDRUG, 128), blk, 0, stream>>>(
                t_y, ldy, Wuse, 128, b1, t_h1, 128, NDRUG, 128, Kg);
            wmma_gemm<1><<<ggrid(NDRUG, 128), blk, 0, stream>>>(
                t_h1, 128, W2, 128, b2, t_x, 128, NDRUG, 128, 128);

            bn_stats_k<<<128, 256, 0, stream>>>(t_x, 128, NDRUG, mu, rs);
            bn_apply_k<<<nblk((long)NDRUG * 128, 256), 256, 0, stream>>>(
                t_x, 128, NDRUG, 128, mu, rs, gg, bb, t_rep + l * 128, 384);

            xin = t_rep + l * 128; ldin = 384; Fin = 128;
        }
        segmax_drug_k<<<nblk(BSZ * 384, 256), 256, 0, stream>>>(t_rep, t_xd);
        wmma_gemm<1><<<ggrid(BSZ, 128), blk, 0, stream>>>(
            t_xd, 384, demb_W, 128, demb_b, xd2, 128, BSZ, 128, 384);
    }

    // ---------------- cell branch: 2x GAT + cluster pool + BN ----------------
    float* xc0 = temp;                       // NC1 x 16 (BN output of GAT0 pool)
    float* gbase = xc0 + (size_t)NC1 * 16;   // GAT scratch (reused by both GATs)

    auto run_gat = [&](const float* x, int K, const float* W, const float* as_,
                       const float* ad_, const float* bias, const int* ei, int E,
                       int n, const int* cluster, int nclus, float* xcout) {
        float*    h   = gbase;
        float*    hs  = h  + (size_t)n * 16;
        float*    hd  = hs + n;
        unsigned* m   = (unsigned*)(hd + n);
        float*    s   = hd + 2 * (size_t)n;
        float*    ev  = s + n;
        float*    out = ev + (size_t)(E + n);
        unsigned* pool = (unsigned*)(out + (size_t)n * 16);

        gat_h_k<<<nblk(n, 256), 256, 0, stream>>>(x, K, W, as_, ad_, h, hs, hd, n);
        hipMemsetAsync(m, 0, (size_t)n * 4, stream);          // enc(-inf) < any real enc
        hipMemsetAsync(s, 0, (size_t)n * 4, stream);
        hipMemsetAsync(out, 0, (size_t)n * 16 * 4, stream);
        hipMemsetAsync(pool, 0, (size_t)nclus * 16 * 4, stream); // 0.0f; relu => max>=0

        int tot = E + n;
        gat_logit_max_k<<<nblk(tot, 256), 256, 0, stream>>>(hs, hd, ei, ei + E, E, n, ev, m);
        gat_exp_sum_k<<<nblk(tot, 256), 256, 0, stream>>>(ev, ei + E, E, n, m, s);
        gat_accum_k<<<nblk((long)tot * 16, 256), 256, 0, stream>>>(ev, h, ei, ei + E, s, out, E, n);
        gat_finish_pool_k<<<nblk((long)n * 16, 256), 256, 0, stream>>>(out, bias, cluster, pool, n);

        bn_stats_k<<<16, 256, 0, stream>>>((const float*)pool, 16, nclus, mu, rs);
        bn_apply_k<<<nblk((long)nclus * 16, 256), 256, 0, stream>>>(
            (const float*)pool, 16, nclus, 16, mu, rs, nullptr, nullptr, xcout, 16);
    };

    run_gat(cell_x, 3,  F(22), F(23), F(24), F(25), ei0, EC0, NCELL, cl0, NC1, xc0);
    run_gat(xc0,    16, F(26), F(27), F(28), F(29), ei1, E1,  NC1,   cl1, NC2, xcf);

    // ---------------- cell embed: (512,6400)@(6400,1024) -> (512,256) ----------------
    wmma_gemm<1><<<ggrid(BSZ, 1024), blk, 0, stream>>>(
        xcf, 6400, cemb_W1, 1024, cemb_b1, a1, 1024, BSZ, 1024, 6400);
    wmma_gemm<1><<<ggrid(BSZ, 256), blk, 0, stream>>>(
        a1, 1024, cemb_W2, 256, cemb_b2, a2, 256, BSZ, 256, 1024);

    // ---------------- regression head ----------------
    concat_k<<<nblk(BSZ * 384, 256), 256, 0, stream>>>(xd2, a2, hcat);
    wmma_gemm<2><<<ggrid(BSZ, 384), blk, 0, stream>>>(
        hcat, 384, reg_W1, 384, reg_b1, r1, 384, BSZ, 384, 384);
    wmma_gemm<2><<<ggrid(BSZ, 384), blk, 0, stream>>>(
        r1, 384, reg_W2, 384, reg_b2, r2, 384, BSZ, 384, 384);
    head_dot_k<<<BSZ, 32, 0, stream>>>(r2, reg_W3, reg_b3, (float*)d_out);
}
